// LightConvolutionLayer_81905026335008
// MI455X (gfx1250) — matrix operand order repaired and checked by reading, not compiled, not run on hardware
//
#include <hip/hip_runtime.h>
#include <hip/hip_bf16.h>

// ---------------------------------------------------------------------------
// LightConvolutionLayer for MI455X (gfx1250)
//   wave32, V_WMMA_F32_16X16X32_BF16, async global->LDS staging (ASYNCcnt)
//   B=8, T=4096, D=1024, H=16, K=15
// Pipeline:
//   k_cvt_x   : X fp32 -> bf16                        (streaming, one pass)
//   k_cvt_wt  : W1,W2 fp32 -> bf16, transposed [n][k] (tiny, L2-resident)
//   k_gemm1   : GLU(Xbf * W1T + b1) -> x bf16         (WMMA, async LDS staging)
//   k_dwconv  : depthwise 15-tap conv along T         (softmax filter in LDS)
//   k_gemm2   : xconv * W2T + b2 -> out fp32          (WMMA, async LDS staging)
// ---------------------------------------------------------------------------

typedef __attribute__((ext_vector_type(16))) __bf16 v16bf;
typedef __attribute__((ext_vector_type(8)))  float  v8f;
typedef unsigned int v4u __attribute__((ext_vector_type(4)));
typedef int          v4i __attribute__((ext_vector_type(4)));
typedef float        v4f __attribute__((ext_vector_type(4)));

#define NB    8
#define NT    4096
#define ND    1024
#define NH    16
#define NK    15
#define MTOT  (NB * NT)   // 32768 rows
#define LDA   48          // LDS row pitch (shorts): 96B, 16B-aligned, spreads banks
#define LDB   48

#if defined(__has_builtin)
#if __has_builtin(__builtin_amdgcn_global_load_async_to_lds_b128)
#define HAVE_ASYNC_LDS 1
#endif
#endif

static __device__ __forceinline__ unsigned short f2bf(float f) {
  unsigned int u = __float_as_uint(f);
  u += 0x7fffu + ((u >> 16) & 1u);           // round-to-nearest-even
  return (unsigned short)(u >> 16);
}
static __device__ __forceinline__ float bf2f(unsigned short h) {
  return __uint_as_float(((unsigned int)h) << 16);
}

// 16B global->LDS copy: async DMA on gfx1250 (ASYNCcnt), else VGPR round-trip.
static __device__ __forceinline__ void cp_b128(const unsigned short* g,
                                               unsigned short* l) {
#ifdef HAVE_ASYNC_LDS
  __builtin_amdgcn_global_load_async_to_lds_b128(
      (__attribute__((address_space(1))) v4i*)(v4i*)g,   // global src (addrspacecast)
      (__attribute__((address_space(3))) v4i*)(v4i*)l,   // LDS dst   (addrspacecast)
      0, 0);
#else
  *(v4u*)l = *(const v4u*)g;
#endif
}

static __device__ __forceinline__ void wait_async() {
#ifdef HAVE_ASYNC_LDS
#if __has_builtin(__builtin_amdgcn_s_wait_asynccnt)
  __builtin_amdgcn_s_wait_asynccnt(0);
#else
  asm volatile("s_wait_asynccnt 0" ::: "memory");
#endif
#endif
}

union FragAB { v16bf v; v4u q[2]; };

// ---------------------------------------------------------------------------
// X fp32 -> bf16, 8 elements/thread, packed 16B stores
// ---------------------------------------------------------------------------
__global__ __launch_bounds__(256)
void k_cvt_x(const float* __restrict__ X, unsigned short* __restrict__ Y) {
  const size_t i = ((size_t)blockIdx.x * 256 + threadIdx.x) * 8;
  const v4f a = *(const v4f*)(X + i);
  const v4f b = *(const v4f*)(X + i + 4);
  v4u p;
  p.x = (unsigned)f2bf(a.x) | ((unsigned)f2bf(a.y) << 16);
  p.y = (unsigned)f2bf(a.z) | ((unsigned)f2bf(a.w) << 16);
  p.z = (unsigned)f2bf(b.x) | ((unsigned)f2bf(b.y) << 16);
  p.w = (unsigned)f2bf(b.z) | ((unsigned)f2bf(b.w) << 16);
  *(v4u*)(Y + i) = p;
}

// ---------------------------------------------------------------------------
// W[K][N] fp32 -> WT[N][K] bf16 (tiled transpose via LDS)
// ---------------------------------------------------------------------------
__global__ __launch_bounds__(256)
void k_cvt_wt(const float* __restrict__ W, unsigned short* __restrict__ WT,
              int K, int N) {
  __shared__ float tile[32][33];
  const int tx = threadIdx.x & 31, ty = threadIdx.x >> 5;   // ty 0..7
  const int n0 = blockIdx.x * 32, k0 = blockIdx.y * 32;
  #pragma unroll
  for (int i = 0; i < 4; ++i)
    tile[ty + i * 8][tx] = W[(size_t)(k0 + ty + i * 8) * N + n0 + tx];
  __syncthreads();
  #pragma unroll
  for (int i = 0; i < 4; ++i)
    WT[(size_t)(n0 + ty + i * 8) * K + k0 + tx] = f2bf(tile[tx][ty + i * 8]);
}

// ---------------------------------------------------------------------------
// Kernel: x = a * sigmoid(g), [a|g] = Xbf(32768x1024) * W1 + b1
// 128x64 tile / 256-thread WG (8 waves, 4x2); per wave 2x2 WMMA tiles for
// both halves. All tile staging is b128 async copies (everything is bf16).
// ---------------------------------------------------------------------------
__global__ __launch_bounds__(256)
void k_gemm1_glu(const unsigned short* __restrict__ Xbf,
                 const unsigned short* __restrict__ W1T,   // [2048 n][1024 k] bf16
                 const float* __restrict__ b1,
                 unsigned short* __restrict__ Y) {
  __shared__ __align__(16) unsigned short sA [128 * LDA];
  __shared__ __align__(16) unsigned short sBa[ 64 * LDB];
  __shared__ __align__(16) unsigned short sBg[ 64 * LDB];

  const int tid  = threadIdx.x;
  const int lane = tid & 31;
  const int wave = tid >> 5;
  const int wM   = (wave & 3) * 32;
  const int wN   = (wave >> 2) * 32;
  const int m0   = blockIdx.y * 128;
  const int n0   = blockIdx.x * 64;
  const int kg   = lane >> 4;
  const int lr   = lane & 15;

  const v8f vz = {0.f,0.f,0.f,0.f,0.f,0.f,0.f,0.f};
  v8f accA[2][2], accG[2][2];
  #pragma unroll
  for (int i = 0; i < 2; ++i)
    #pragma unroll
    for (int j = 0; j < 2; ++j) { accA[i][j] = vz; accG[i][j] = vz; }

  // per-thread staging slots (fixed across K iterations)
  const int ar0 = tid >> 2,         as0 = (tid & 3) * 8;          // A seg 1
  const int ar1 = (tid + 256) >> 2, as1 = ((tid + 256) & 3) * 8;  // A seg 2
  const int bn  = tid >> 2,         bs  = (tid & 3) * 8;          // B seg

  for (int k0 = 0; k0 < ND; k0 += 32) {
    cp_b128(Xbf + (size_t)(m0 + ar0) * ND + k0 + as0, sA + ar0 * LDA + as0);
    cp_b128(Xbf + (size_t)(m0 + ar1) * ND + k0 + as1, sA + ar1 * LDA + as1);
    cp_b128(W1T + (size_t)(n0 + bn) * ND + k0 + bs,        sBa + bn * LDB + bs);
    cp_b128(W1T + (size_t)(1024 + n0 + bn) * ND + k0 + bs, sBg + bn * LDB + bs);
    if (k0 + 32 < ND)  // gfx1250 global_prefetch_b8 for next K tile
      __builtin_prefetch(Xbf + (size_t)(m0 + (tid >> 1)) * ND + k0 + 32, 0, 0);
    wait_async();
    __syncthreads();

    // fragments per ISA 7.12.2 layouts
    FragAB aF[2], bA[2], bG[2];
    #pragma unroll
    for (int mi = 0; mi < 2; ++mi) {
      const unsigned short* p = sA + (wM + mi * 16 + lr) * LDA + kg * 8;
      aF[mi].q[0] = *(const v4u*)p;          // K = kg*8 .. +7
      aF[mi].q[1] = *(const v4u*)(p + 16);   // K = 16+kg*8 .. +7
    }
    #pragma unroll
    for (int ni = 0; ni < 2; ++ni) {
      const unsigned short* pa = sBa + (wN + ni * 16 + lr) * LDB + kg * 8;
      bA[ni].q[0] = *(const v4u*)pa;
      bA[ni].q[1] = *(const v4u*)(pa + 16);
      const unsigned short* pg = sBg + (wN + ni * 16 + lr) * LDB + kg * 8;
      bG[ni].q[0] = *(const v4u*)pg;
      bG[ni].q[1] = *(const v4u*)(pg + 16);
    }
    #pragma unroll
    for (int mi = 0; mi < 2; ++mi)
      #pragma unroll
      for (int ni = 0; ni < 2; ++ni) {
        accA[mi][ni] = __builtin_amdgcn_wmma_f32_16x16x32_bf16(
            false, aF[mi].v, false, bA[ni].v, (short)0, accA[mi][ni], false, false);
        accG[mi][ni] = __builtin_amdgcn_wmma_f32_16x16x32_bf16(
            false, aF[mi].v, false, bG[ni].v, (short)0, accG[mi][ni], false, false);
      }
    __syncthreads();
  }

  #pragma unroll
  for (int mi = 0; mi < 2; ++mi)
    #pragma unroll
    for (int ni = 0; ni < 2; ++ni) {
      const int n = n0 + wN + ni * 16 + lr;
      const float ba = b1[n];
      const float bg = b1[ND + n];
      #pragma unroll
      for (int j = 0; j < 8; ++j) {
        const int m = m0 + wM + mi * 16 + kg * 8 + j;  // C/D: M=(L>>4)*8+j, N=L&15
        const float av = accA[mi][ni][j] + ba;
        const float gv = accG[mi][ni][j] + bg;
        const float xv = av * (1.f / (1.f + __expf(-gv)));
        Y[(size_t)m * ND + n] = f2bf(xv);
      }
    }
}

// ---------------------------------------------------------------------------
// Depthwise 15-tap conv along T, SAME padding, softmax(16x15) filter in LDS.
// ---------------------------------------------------------------------------
__global__ __launch_bounds__(256)
void k_dwconv(const unsigned short* __restrict__ X, const float* __restrict__ Wc,
              unsigned short* __restrict__ Y) {
  __shared__ float filt[NH][NK];
  const int tid = threadIdx.x;
  if (tid < NH) {
    float mx = -__builtin_inff();
    float e[NK];
    #pragma unroll
    for (int k = 0; k < NK; ++k) mx = fmaxf(mx, Wc[tid * NK + k]);
    float s = 0.f;
    #pragma unroll
    for (int k = 0; k < NK; ++k) { e[k] = __expf(Wc[tid * NK + k] - mx); s += e[k]; }
    const float inv = 1.f / s;
    #pragma unroll
    for (int k = 0; k < NK; ++k) filt[tid][k] = e[k] * inv;
  }
  __syncthreads();

  const int d = blockIdx.x * 256 + tid;
  const int t = blockIdx.y;
  const int b = blockIdx.z;
  const float* f = filt[d >> 6];
  const size_t rowBase = ((size_t)b * NT) * ND + d;

  float acc = 0.f;
  #pragma unroll
  for (int k = 0; k < NK; ++k) {
    const int tt = t + k - (NK / 2);
    if (tt >= 0 && tt < NT)
      acc += bf2f(X[rowBase + (size_t)tt * ND]) * f[k];
  }
  Y[rowBase + (size_t)t * ND] = f2bf(acc);
}

// ---------------------------------------------------------------------------
// Out = Xc(bf16) * W2 + b2, fp32 output.  W2T is [1024 n][1024 k] bf16.
// ---------------------------------------------------------------------------
__global__ __launch_bounds__(256)
void k_gemm2(const unsigned short* __restrict__ Xc,
             const unsigned short* __restrict__ W2T,
             const float* __restrict__ b2, float* __restrict__ Out) {
  __shared__ __align__(16) unsigned short sA[128 * LDA];
  __shared__ __align__(16) unsigned short sB[ 64 * LDB];

  const int tid  = threadIdx.x;
  const int lane = tid & 31;
  const int wave = tid >> 5;
  const int wM   = (wave & 3) * 32;
  const int wN   = (wave >> 2) * 32;
  const int m0   = blockIdx.y * 128;
  const int n0   = blockIdx.x * 64;
  const int kg   = lane >> 4;
  const int lr   = lane & 15;

  const v8f vz = {0.f,0.f,0.f,0.f,0.f,0.f,0.f,0.f};
  v8f acc[2][2];
  #pragma unroll
  for (int i = 0; i < 2; ++i)
    #pragma unroll
    for (int j = 0; j < 2; ++j) acc[i][j] = vz;

  const int ar0 = tid >> 2,         as0 = (tid & 3) * 8;
  const int ar1 = (tid + 256) >> 2, as1 = ((tid + 256) & 3) * 8;
  const int bn  = tid >> 2,         bs  = (tid & 3) * 8;

  for (int k0 = 0; k0 < ND; k0 += 32) {
    cp_b128(Xc  + (size_t)(m0 + ar0) * ND + k0 + as0, sA + ar0 * LDA + as0);
    cp_b128(Xc  + (size_t)(m0 + ar1) * ND + k0 + as1, sA + ar1 * LDA + as1);
    cp_b128(W2T + (size_t)(n0 + bn) * ND + k0 + bs,   sB + bn * LDB + bs);
    if (k0 + 32 < ND)
      __builtin_prefetch(Xc + (size_t)(m0 + (tid >> 1)) * ND + k0 + 32, 0, 0);
    wait_async();
    __syncthreads();

    FragAB aF[2], bF[2];
    #pragma unroll
    for (int mi = 0; mi < 2; ++mi) {
      const unsigned short* p = sA + (wM + mi * 16 + lr) * LDA + kg * 8;
      aF[mi].q[0] = *(const v4u*)p;
      aF[mi].q[1] = *(const v4u*)(p + 16);
    }
    #pragma unroll
    for (int ni = 0; ni < 2; ++ni) {
      const unsigned short* p = sB + (wN + ni * 16 + lr) * LDB + kg * 8;
      bF[ni].q[0] = *(const v4u*)p;
      bF[ni].q[1] = *(const v4u*)(p + 16);
    }
    #pragma unroll
    for (int mi = 0; mi < 2; ++mi)
      #pragma unroll
      for (int ni = 0; ni < 2; ++ni)
        acc[mi][ni] = __builtin_amdgcn_wmma_f32_16x16x32_bf16(
            false, aF[mi].v, false, bF[ni].v, (short)0, acc[mi][ni], false, false);
    __syncthreads();
  }

  #pragma unroll
  for (int mi = 0; mi < 2; ++mi)
    #pragma unroll
    for (int ni = 0; ni < 2; ++ni) {
      const int n = n0 + wN + ni * 16 + lr;
      const float bb = b2[n];
      #pragma unroll
      for (int j = 0; j < 8; ++j) {
        const int m = m0 + wM + mi * 16 + kg * 8 + j;
        Out[(size_t)m * ND + n] = acc[mi][ni][j] + bb;
      }
    }
}

// ---------------------------------------------------------------------------
extern "C" void kernel_launch(void* const* d_in, const int* in_sizes, int n_in,
                              void* d_out, int out_size, void* d_ws, size_t ws_size,
                              hipStream_t stream) {
  (void)in_sizes; (void)n_in; (void)out_size; (void)ws_size;
  const float* X  = (const float*)d_in[0];   // (8,4096,1024)
  const float* W1 = (const float*)d_in[1];   // (1024,2048)
  const float* b1 = (const float*)d_in[2];   // (2048,)
  const float* W2 = (const float*)d_in[3];   // (1024,1024)
  const float* b2 = (const float*)d_in[4];   // (1024,)
  const float* Wc = (const float*)d_in[5];   // (16,15)
  float* Out = (float*)d_out;

  // workspace layout (xconv aliases xbf: xbf is dead after k_gemm1_glu)
  unsigned short* xbf   = (unsigned short*)d_ws;                  // 64 MB
  unsigned short* xglu  = xbf + (size_t)MTOT * ND;                // 64 MB
  unsigned short* xconv = xbf;                                    // alias
  unsigned short* w1t   = xglu + (size_t)MTOT * ND;               // 4 MB
  unsigned short* w2t   = w1t + (size_t)(2 * ND) * ND;            // 2 MB

  k_cvt_x<<<dim3((MTOT * ND) / (256 * 8)), 256, 0, stream>>>(X, xbf);
  k_cvt_wt<<<dim3((2 * ND) / 32, ND / 32), 256, 0, stream>>>(W1, w1t, ND, 2 * ND);
  k_cvt_wt<<<dim3(ND / 32, ND / 32), 256, 0, stream>>>(W2, w2t, ND, ND);

  dim3 gGemm(ND / 64, MTOT / 128);  // (16, 256)
  k_gemm1_glu<<<gGemm, 256, 0, stream>>>(xbf, w1t, b1, xglu);

  dim3 gConv(ND / 256, NT, NB);     // (4, 4096, 8)
  k_dwconv<<<gConv, 256, 0, stream>>>(xglu, Wc, xconv);

  k_gemm2<<<gGemm, 256, 0, stream>>>(xconv, w2t, b2, Out);
}
